// Attention_1683627180818
// MI455X (gfx1250) — matrix operand order repaired and checked by reading, not compiled, not run on hardware
//
#include <hip/hip_runtime.h>
#include <hip/hip_bf16.h>

typedef __bf16 bf16_t;
typedef __attribute__((ext_vector_type(16))) __bf16 v16bf;
typedef __attribute__((ext_vector_type(8)))  __bf16 v8bf;
typedef __attribute__((ext_vector_type(4)))  __bf16 v4bf;
typedef __attribute__((ext_vector_type(8)))  float  v8f;
typedef __attribute__((ext_vector_type(4)))  float  v4f;
typedef __attribute__((ext_vector_type(4)))  unsigned int u32x4;
typedef __attribute__((ext_vector_type(8)))  int i32x8;
typedef __attribute__((ext_vector_type(4)))  int i32x4;

#define DIMD   4096
#define NH     32
#define NKV    8
#define HD     128
#define SEQL   2048
#define BATCH  2
#define OPSZ   6144
#define QPOS   4096
#define KVSZ   1024

// Native bf16 converts (v_cvt on gfx1250, RNE)
__device__ __forceinline__ bf16_t f2bf(float f) { return (bf16_t)f; }
__device__ __forceinline__ float  bf2f(bf16_t b) { return (float)b; }
__device__ __forceinline__ void store_c(float* p, float v)  { *p = v; }
__device__ __forceinline__ void store_c(bf16_t* p, float v) { *p = f2bf(v); }

// =====================================================================
// TDM: async 2-D bf16 tile load Global -> LDS (D# packed per ISA §8.3/8.4)
// amdgpu-toolchain (clang-23) 6-arg builtin form.
// =====================================================================
__device__ __forceinline__ void tdm_load_2d(unsigned lds_off, const bf16_t* gaddr,
                                            int rows_total, int cols_total,
                                            int tile_rows, int tile_cols)
{
  unsigned long long ga = (unsigned long long)(size_t)gaddr;
  u32x4 g0;
  g0[0] = 1u;                                           // count=1, user mode
  g0[1] = lds_off;                                      // lds_addr (bytes)
  g0[2] = (unsigned)(ga & 0xFFFFFFFFu);                 // global_addr[31:0]
  g0[3] = (unsigned)((ga >> 32) & 0x01FFFFFFu) | (2u << 30);  // addr[56:32] | type=2
  i32x8 g1;
  g1[0] = (int)(1u << 16);                              // wg_mask=0, data_size=1 (2B)
  g1[1] = (int)(((unsigned)cols_total & 0xFFFFu) << 16);           // tensor_dim0 lo16
  g1[2] = (int)((((unsigned)cols_total >> 16) & 0xFFFFu)
              | (((unsigned)rows_total & 0xFFFFu) << 16));         // dim0 hi | dim1 lo
  g1[3] = (int)((((unsigned)rows_total >> 16) & 0xFFFFu)
              | (((unsigned)tile_cols & 0xFFFFu) << 16));          // dim1 hi | tile_dim0
  g1[4] = (int)((unsigned)tile_rows & 0xFFFFu);                    // tile_dim1, tile_dim2=0
  g1[5] = (int)(unsigned)cols_total;                               // tensor_dim0_stride lo32
  g1[6] = 0;
  g1[7] = 0;
  i32x4 z4 = {0, 0, 0, 0};
  i32x8 z8 = {0, 0, 0, 0, 0, 0, 0, 0};
  __builtin_amdgcn_tensor_load_to_lds(g0, g1, z4, z4, z8, 0);
}

// =====================================================================
// One-pass converts: x -> bf16, and W[K][N] f32 -> WT[N][K] bf16 (tiled)
// =====================================================================
__global__ __launch_bounds__(256)
void convert_f32_bf16(const float* __restrict__ in, bf16_t* __restrict__ out)
{
  const size_t i = ((size_t)blockIdx.x * 256 + threadIdx.x) * 4;
  v4f v = *(const v4f*)(in + i);
  v4bf o;
#pragma unroll
  for (int j = 0; j < 4; ++j) o[j] = f2bf(v[j]);
  *(v4bf*)(out + i) = o;
}

__global__ __launch_bounds__(256)
void transpose_f32_to_bf16(const float* __restrict__ W, bf16_t* __restrict__ WT,
                           int K, int N)   // W[K][N] -> WT[N][K]
{
  __shared__ float tile[32][33];
  const int tx = threadIdx.x & 31;
  const int ty = threadIdx.x >> 5;          // 0..7
  const int k0 = blockIdx.y * 32;
  const int n0 = blockIdx.x * 32;
#pragma unroll
  for (int i = 0; i < 4; ++i)
    tile[ty * 4 + i][tx] = W[(size_t)(k0 + ty * 4 + i) * N + n0 + tx];
  __syncthreads();
#pragma unroll
  for (int i = 0; i < 4; ++i)
    WT[(size_t)(n0 + ty * 4 + i) * K + k0 + tx] = f2bf(tile[tx][ty * 4 + i]);
}

// =====================================================================
// GEMM: C[M,N] = A[M,K](bf16) @ WT[N,K](bf16, pre-transposed)
// 128x128 block, K-step 64, DOUBLE-BUFFERED TDM pipeline:
//   prologue: wave0 DMAs tile 0
//   iter i : wave0 DMAs tile i+1 into buf^1, waits tensorcnt<=2
//            (TDM completes in order -> tile i landed), all waves compute.
// 8 waves: 2(m) x 4(n); each wave 4x2 subtiles x 2 k-chunks = 16 WMMAs/fill.
// =====================================================================
template<typename CT>
__global__ __launch_bounds__(256)
void gemm_wmma_tdm(const bf16_t* __restrict__ A, const bf16_t* __restrict__ WT,
                   CT* __restrict__ C, int M, int N, int K)
{
  __shared__ bf16_t sA[2][128 * 64];   // [buf][m][k]
  __shared__ bf16_t sB[2][128 * 64];   // [buf][n][k]
  const int tid  = threadIdx.x;
  const int lane = tid & 31;
  const int wave = tid >> 5;
  const int half = lane >> 4;
  const int l16  = lane & 15;
  const int wm   = wave & 1;
  const int wn   = wave >> 1;
  const int bm   = blockIdx.y * 128;
  const int bn   = blockIdx.x * 128;

  v8f acc[4][2];
#pragma unroll
  for (int i = 0; i < 4; ++i)
#pragma unroll
    for (int j = 0; j < 2; ++j)
#pragma unroll
      for (int r = 0; r < 8; ++r) acc[i][j][r] = 0.0f;

  const bf16_t* Abase = A  + (size_t)bm * K;
  const bf16_t* Bbase = WT + (size_t)bn * K;

  if (wave == 0) {
    tdm_load_2d((unsigned)(size_t)&sA[0][0], Abase, M, K, 128, 64);
    tdm_load_2d((unsigned)(size_t)&sB[0][0], Bbase, N, K, 128, 64);
  }

  int buf = 0;
  for (int k0 = 0; k0 < K; k0 += 64) {
    if (wave == 0) {
      if (k0 + 64 < K) {
        tdm_load_2d((unsigned)(size_t)&sA[buf ^ 1][0], Abase + k0 + 64, M, K, 128, 64);
        tdm_load_2d((unsigned)(size_t)&sB[buf ^ 1][0], Bbase + k0 + 64, N, K, 128, 64);
        __builtin_amdgcn_s_wait_tensorcnt(2);   // tile k0 complete; next tile in flight
      } else {
        __builtin_amdgcn_s_wait_tensorcnt(0);
      }
    }
    __syncthreads();

    const bf16_t* a = &sA[buf][0];
    const bf16_t* b = &sB[buf][0];
#pragma unroll
    for (int kc = 0; kc < 2; ++kc) {
      v16bf afrag[4];
#pragma unroll
      for (int im = 0; im < 4; ++im) {
        const int m = wm * 64 + im * 16 + l16;
        union { v16bf v; v8bf h[2]; } u;
        u.h[0] = *(const v8bf*)&a[m * 64 + kc * 32 + half * 16];
        u.h[1] = *(const v8bf*)&a[m * 64 + kc * 32 + half * 16 + 8];
        afrag[im] = u.v;
      }
#pragma unroll
      for (int in = 0; in < 2; ++in) {
        const int n = wn * 32 + in * 16 + l16;
        v16bf bfrag = *(const v16bf*)&b[n * 64 + kc * 32 + half * 16];
#pragma unroll
        for (int im = 0; im < 4; ++im)
          acc[im][in] = __builtin_amdgcn_wmma_f32_16x16x32_bf16(
              false, afrag[im], false, bfrag, (short)0, acc[im][in], false, false);
      }
    }
    __syncthreads();
    buf ^= 1;
  }

#pragma unroll
  for (int im = 0; im < 4; ++im)
#pragma unroll
    for (int in = 0; in < 2; ++in) {
      const int col = bn + wn * 32 + in * 16 + l16;
#pragma unroll
      for (int r = 0; r < 8; ++r) {
        const int row = bm + wm * 64 + im * 16 + half * 8 + r;
        store_c(C + (size_t)row * N + col, acc[im][in][r]);
      }
    }
}

// =====================================================================
// RoPE (Q in place) + K -> cache[b][kh][s][d] + V -> vt[b][kh][d][s]
// =====================================================================
__global__ __launch_bounds__(256)
void rope_cache_kernel(bf16_t* __restrict__ qkv, const float* __restrict__ sincos,
                       bf16_t* __restrict__ kc, bf16_t* __restrict__ vt)
{
  const int gid = blockIdx.x * 256 + threadIdx.x;  // B*S*48*64
  const int d   = gid & 63;
  const int hh  = (gid >> 6) % 48;
  const int tok = gid / (48 * 64);
  const int s   = tok & (SEQL - 1);
  const int b   = tok >> 11;
  bf16_t* row = qkv + (size_t)tok * OPSZ;
  if (hh < 40) {
    const float sn = sincos[s * HD + d];
    const float cs = sincos[s * HD + 64 + d];
    if (hh < 32) {
      bf16_t* p = row + hh * HD + d;
      float u0 = bf2f(p[0]), u1 = bf2f(p[64]);
      p[0]  = f2bf(u0 * cs - u1 * sn);
      p[64] = f2bf(u1 * cs + u0 * sn);
    } else {
      const int kh = hh - 32;
      const bf16_t* p = row + QPOS + kh * HD + d;
      float u0 = bf2f(p[0]), u1 = bf2f(p[64]);
      bf16_t* o = kc + (((size_t)(b * NKV + kh) * SEQL + s) * HD + d);
      o[0]  = f2bf(u0 * cs - u1 * sn);
      o[64] = f2bf(u1 * cs + u0 * sn);
    }
  } else {
    const int vh = hh - 40;
    const bf16_t* p = row + QPOS + KVSZ + vh * HD + d;
    bf16_t* o = vt + ((size_t)(b * NKV + vh) * HD + d) * SEQL + s;
    o[0]                 = p[0];
    o[(size_t)64 * SEQL] = p[64];
  }
}

// =====================================================================
// Flash attention: one wave per (b, h, 16-row q block). GQA kh = h/4.
// =====================================================================
__global__ __launch_bounds__(256)
void attn_wmma_kernel(const bf16_t* __restrict__ qkv, const bf16_t* __restrict__ kc,
                      const bf16_t* __restrict__ vt, bf16_t* __restrict__ aout)
{
  __shared__ bf16_t sP[8][16 * 32];
  const int lane = threadIdx.x & 31;
  const int wave = threadIdx.x >> 5;
  const int half = lane >> 4;
  const int l16  = lane & 15;
  const int task = blockIdx.x * 8 + wave;   // B*NH*(SEQ/16) = 8192
  const int qb   = task & 127;
  const int h    = (task >> 7) & 31;
  const int b    = task >> 12;
  const int q0   = qb * 16;
  const int kh   = h >> 2;
  bf16_t* myP = sP[wave];

  const bf16_t* qbase = qkv + ((size_t)(b * SEQL + q0 + l16) * OPSZ) + h * HD;
  v16bf qfrag[4];
#pragma unroll
  for (int kc4 = 0; kc4 < 4; ++kc4) {
    union { v16bf v; v8bf hv[2]; } u;
    u.hv[0] = *(const v8bf*)(qbase + kc4 * 32 + half * 16);
    u.hv[1] = *(const v8bf*)(qbase + kc4 * 32 + half * 16 + 8);
    qfrag[kc4] = u.v;
  }

  v8f oacc[8];
#pragma unroll
  for (int j = 0; j < 8; ++j)
#pragma unroll
    for (int r = 0; r < 8; ++r) oacc[j][r] = 0.0f;
  float rmax[8], rsum[8];
#pragma unroll
  for (int r = 0; r < 8; ++r) { rmax[r] = -1e30f; rsum[r] = 0.0f; }

  const float scale = 0.08838834764831845f;  // 1/sqrt(128)
  const bf16_t* kbase = kc + (size_t)(b * NKV + kh) * SEQL * HD;
  const bf16_t* vbase = vt + (size_t)(b * NKV + kh) * HD * SEQL;

  const int kvlim = q0 + 16;
  for (int kvb = 0; kvb < kvlim; kvb += 32) {
    v8f sc[2];
#pragma unroll
    for (int sub = 0; sub < 2; ++sub)
#pragma unroll
      for (int r = 0; r < 8; ++r) sc[sub][r] = 0.0f;

#pragma unroll
    for (int sub = 0; sub < 2; ++sub) {
      const int kv = kvb + sub * 16 + l16;
#pragma unroll
      for (int kc4 = 0; kc4 < 4; ++kc4) {
        v16bf kf = *(const v16bf*)(kbase + (size_t)kv * HD + kc4 * 32 + half * 16);
        sc[sub] = __builtin_amdgcn_wmma_f32_16x16x32_bf16(
            false, qfrag[kc4], false, kf, (short)0, sc[sub], false, false);
      }
    }

    float p0[8], p1[8];
#pragma unroll
    for (int r = 0; r < 8; ++r) {
      const int row  = q0 + half * 8 + r;
      const int col0 = kvb + l16;
      const int col1 = kvb + 16 + l16;
      float v0 = sc[0][r] * scale, v1 = sc[1][r] * scale;
      v0 = (col0 <= row) ? v0 : -1e30f;
      v1 = (col1 <= row) ? v1 : -1e30f;
      float m = fmaxf(v0, v1);
      m = fmaxf(m, __shfl_xor(m, 1, 32));
      m = fmaxf(m, __shfl_xor(m, 2, 32));
      m = fmaxf(m, __shfl_xor(m, 4, 32));
      m = fmaxf(m, __shfl_xor(m, 8, 32));
      const float nm   = fmaxf(rmax[r], m);
      const float corr = __expf(rmax[r] - nm);
      rmax[r] = nm;
      rsum[r] *= corr;
#pragma unroll
      for (int j = 0; j < 8; ++j) oacc[j][r] *= corr;
      p0[r] = __expf(v0 - nm);
      p1[r] = __expf(v1 - nm);
      float ps = p0[r] + p1[r];
      ps += __shfl_xor(ps, 1, 32);
      ps += __shfl_xor(ps, 2, 32);
      ps += __shfl_xor(ps, 4, 32);
      ps += __shfl_xor(ps, 8, 32);
      rsum[r] += ps;
    }

#pragma unroll
    for (int r = 0; r < 8; ++r) {
      myP[(half * 8 + r) * 32 + l16]      = f2bf(p0[r]);
      myP[(half * 8 + r) * 32 + 16 + l16] = f2bf(p1[r]);
    }
    asm volatile("s_wait_dscnt 0" ::: "memory");
    union { v16bf v; v8bf hv[2]; } up;
    up.hv[0] = *(const v8bf*)&myP[l16 * 32 + half * 16];
    up.hv[1] = *(const v8bf*)&myP[l16 * 32 + half * 16 + 8];
    v16bf pfrag = up.v;

#pragma unroll
    for (int j = 0; j < 8; ++j) {
      v16bf vf = *(const v16bf*)(vbase + (size_t)(j * 16 + l16) * SEQL + kvb + half * 16);
      oacc[j] = __builtin_amdgcn_wmma_f32_16x16x32_bf16(
          false, pfrag, false, vf, (short)0, oacc[j], false, false);
    }
  }

#pragma unroll
  for (int j = 0; j < 8; ++j)
#pragma unroll
    for (int r = 0; r < 8; ++r) {
      const int row = q0 + half * 8 + r;
      aout[((size_t)(b * SEQL + row) * DIMD) + h * HD + j * 16 + l16] =
          f2bf(oacc[j][r] / rsum[r]);
    }
}

// =====================================================================
// Host-side orchestration
// =====================================================================
extern "C" void kernel_launch(void* const* d_in, const int* in_sizes, int n_in,
                              void* d_out, int out_size, void* d_ws, size_t ws_size,
                              hipStream_t stream)
{
  const float* x      = (const float*)d_in[0];
  const float* wqkv   = (const float*)d_in[1];
  const float* wo     = (const float*)d_in[2];
  const float* sincos = (const float*)d_in[3];
  float* out = (float*)d_out;

  char* ws = (char*)d_ws;
  bf16_t* xb     = (bf16_t*)(ws);                   // 4096*4096*2 = 33554432
  bf16_t* aoutb  = (bf16_t*)(ws);                   // aliases xb (xb dead after GEMM1)
  bf16_t* wqkvT  = (bf16_t*)(ws + 33554432);        // 6144*4096*2 = 50331648
  bf16_t* woT    = (bf16_t*)(ws + 83886080);        // 4096*4096*2 = 33554432
  bf16_t* qkv    = (bf16_t*)(ws + 117440512);       // 4096*6144*2 = 50331648
  bf16_t* kcache = (bf16_t*)(ws + 167772160);       // 8388608
  bf16_t* vtbuf  = (bf16_t*)(ws + 176160768);       // 8388608

  dim3 blk(256);
  convert_f32_bf16<<<dim3((BATCH * SEQL * DIMD) / 1024), blk, 0, stream>>>(x, xb);
  transpose_f32_to_bf16<<<dim3(OPSZ / 32, DIMD / 32), blk, 0, stream>>>(wqkv, wqkvT, DIMD, OPSZ);
  transpose_f32_to_bf16<<<dim3(DIMD / 32, DIMD / 32), blk, 0, stream>>>(wo, woT, DIMD, DIMD);

  gemm_wmma_tdm<bf16_t>
      <<<dim3(OPSZ / 128, (BATCH * SEQL) / 128), blk, 0, stream>>>(
          xb, wqkvT, qkv, BATCH * SEQL, OPSZ, DIMD);

  rope_cache_kernel
      <<<dim3((BATCH * SEQL * 48 * 64) / 256), blk, 0, stream>>>(
          qkv, sincos, kcache, vtbuf);

  attn_wmma_kernel
      <<<dim3((BATCH * NH * (SEQL / 16)) / 8), blk, 0, stream>>>(
          qkv, kcache, vtbuf, aoutb);

  gemm_wmma_tdm<float>
      <<<dim3(DIMD / 128, (BATCH * SEQL) / 128), blk, 0, stream>>>(
          aoutb, woT, out, BATCH * SEQL, DIMD, DIMD);
}
